// TemporalAttentionPaper_13889924235514
// MI455X (gfx1250) — matrix operand order repaired and checked by reading, not compiled, not run on hardware
//
#include <hip/hip_runtime.h>
#include <hip/hip_bf16.h>
#include <math.h>

// Problem constants (from reference)
#define BB 64
#define TT 2048
#define HH 512
#define UU 512

typedef __attribute__((ext_vector_type(16))) _Float16 v16h;
typedef __attribute__((ext_vector_type(8)))  _Float16 v8h;
typedef __attribute__((ext_vector_type(4)))  _Float16 v4h;
typedef __attribute__((ext_vector_type(8)))  float    v8f;

// Explicit LDS (address_space(3)) types so fragment traffic lowers to ds_* ops
typedef __attribute__((address_space(3))) char     lc;
typedef __attribute__((address_space(3))) _Float16 lh;
typedef __attribute__((address_space(3))) v8h      lv8h;
typedef __attribute__((address_space(3))) v4h      lv4h;

__device__ __forceinline__ float tanh_fast(float x) {
#if __has_builtin(__builtin_amdgcn_tanhf)
    return __builtin_amdgcn_tanhf(x);
#else
    return tanhf(x);
#endif
}

// One async global->LDS 128-bit copy per lane (ASYNCcnt-tracked).
__device__ __forceinline__ void async_b128(unsigned lds_off, const void* g) {
    asm volatile("global_load_async_to_lds_b128 %0, %1, off"
                 :: "v"(lds_off), "v"((unsigned long long)(size_t)g)
                 : "memory");
}

__device__ __forceinline__ void wait_async0() {
    asm volatile("s_wait_asynccnt 0" ::: "memory");
}

// ---------------------------------------------------------------------------
// Kernel 1: pqb[b,u] = (concat(query,cellstate)[b,:] @ W1)[u] + b1[u] + b2[u]
// ---------------------------------------------------------------------------
__global__ void k_projq(const float* __restrict__ query,
                        const float* __restrict__ cellstate,
                        const float* __restrict__ W1,
                        const float* __restrict__ b1,
                        const float* __restrict__ b2,
                        float* __restrict__ pqb) {
    __shared__ float sqc[2 * HH];
    const int b = blockIdx.x;
    const int u = threadIdx.x;          // 0..511
    sqc[u]       = query[b * HH + u];
    sqc[HH + u]  = cellstate[b * HH + u];
    __syncthreads();
    float acc = b1[u] + b2[u];
    #pragma unroll 8
    for (int k = 0; k < 2 * HH; ++k)
        acc += sqc[k] * W1[k * UU + u];   // coalesced over u
    pqb[b * UU + u] = acc;
}

// ---------------------------------------------------------------------------
// Kernel 2: W2t[n][k] = (f16) W2[k][n]   (fragment-friendly transposed f16)
// ---------------------------------------------------------------------------
__global__ void k_packW2(const float* __restrict__ W2, _Float16* __restrict__ W2t) {
    const int idx = blockIdx.x * blockDim.x + threadIdx.x; // 0..512*512-1
    const int n = idx >> 9;
    const int k = idx & 511;         // contiguous writes over k
    W2t[n * HH + k] = (_Float16)W2[k * UU + n];
}

// ---------------------------------------------------------------------------
// Kernel 3: fused GEMM (values @ W2) + tanh + dot(V)  ->  raw scores
//   WG = 512 threads (16 waves): 64 t-rows x full U=512 for one b.
//   Wave w: M-tile mt = w>>2 (16 rows), N-chunk wn = w&3 (8 N-tiles).
//   W2 K-slabs (32x512 f16 = 32KB) double-buffered in LDS via
//   global_load_async_to_lds_b128 + s_wait_asynccnt; fixed buffer roles
//   (even slab -> buf0, odd slab -> buf1) to keep the loop straight-line.
//   B fragments are loaded in batches of 4 so WMMAs issue back-to-back.
// ---------------------------------------------------------------------------
#define PITCH   520                       // halves per values row (1040 B)
#define BPITCH  40                        // halves per B-slab column (80 B)
#define SV_BYTES (64 * PITCH * 2)         // 66560
#define SB_BYTES (UU * BPITCH * 2)        // 40960 per buffer
#define SMEM_K3 (SV_BYTES + 2 * SB_BYTES + 64 * 4)

// One K-step: wait for slab KS, barrier, optionally issue slab KS+1 into the
// other buffer, then A-frag ds_load + 8 WMMAs against slab KS in SBX.
#define PROC_HALF(KS_EXPR, SBX, DO_ISSUE, LDSB_NEXT)                          \
  {                                                                           \
    const int ks_ = (KS_EXPR);                                                \
    const int k0_ = ks_ * 32;                                                 \
    wait_async0();            /* this wave's slab(ks) copy complete */        \
    __syncthreads();          /* slab visible; buffer-to-overwrite free */    \
    if (DO_ISSUE) {                                                           \
      const char* src_ = (const char*)gB + (ks_ + 1) * 64;                    \
      _Pragma("unroll")                                                       \
      for (int c = 0; c < 4; ++c)                                             \
        async_b128((LDSB_NEXT) + c * 16, src_ + c * 16);                      \
    }                                                                         \
    const int ka_ = k0_ + laneHi * 8;                                         \
    v8h alo_ = *(const lv8h*)(aBase + ka_);                                   \
    v8h ahi_ = *(const lv8h*)(aBase + ka_ + 16);                              \
    v16h a_ = __builtin_shufflevector(alo_, ahi_,                             \
              0,1,2,3,4,5,6,7,8,9,10,11,12,13,14,15);                         \
    _Pragma("unroll")                                                         \
    for (int g = 0; g < 2; ++g) {                                             \
      v16h bf_[4];                                                            \
      _Pragma("unroll")                                                       \
      for (int q = 0; q < 4; ++q) {                                           \
        const int nt = g * 4 + q;                                             \
        const lh* bp_ = (SBX) + (wn*128 + nt*16 + ln) * BPITCH + laneHi*8;    \
        v8h blo_ = *(const lv8h*)(bp_);                                       \
        v8h bhi_ = *(const lv8h*)(bp_ + 16);                                  \
        bf_[q] = __builtin_shufflevector(blo_, bhi_,                          \
                 0,1,2,3,4,5,6,7,8,9,10,11,12,13,14,15);                      \
      }                                                                       \
      _Pragma("unroll")                                                       \
      for (int q = 0; q < 4; ++q)                                             \
        acc[g * 4 + q] = __builtin_amdgcn_wmma_f32_16x16x32_f16(              \
            false, a_, false, bf_[q], (short)0, acc[g * 4 + q], false, false);\
    }                                                                         \
  }

__global__ __launch_bounds__(512) void k_score(
        const float* __restrict__ values,
        const _Float16* __restrict__ W2t,
        const float* __restrict__ pqb,
        const float* __restrict__ Vv,
        float* __restrict__ scoreWS) {
    extern __shared__ char smem[];
    lc* lbase = (lc*)smem;                              // LDS address space
    lh* sV    = (lh*)lbase;                             // [64][PITCH] f16
    lh* sB0   = (lh*)(lbase + SV_BYTES);                // B slab buffer 0
    lh* sB1   = (lh*)(lbase + SV_BYTES + SB_BYTES);     // B slab buffer 1
    float* sScorG = (float*)(smem + SV_BYTES + 2 * SB_BYTES); // generic (atomics)

    const int b   = blockIdx.y;
    const int t0  = blockIdx.x * 64;
    const int tid = threadIdx.x;

    if (tid < 64) sScorG[tid] = 0.0f;

    // ---- B-slab async copy: thread tid owns column n = tid (64 B/slab) ----
    const _Float16* gB = W2t + (size_t)tid * HH;
    const unsigned ldsB0 = (unsigned)(size_t)sB0 + (unsigned)tid * (BPITCH * 2);
    const unsigned ldsB1 = (unsigned)(size_t)sB1 + (unsigned)tid * (BPITCH * 2);

    // prologue: slab 0 (k = 0..31) -> buffer 0
    #pragma unroll
    for (int c = 0; c < 4; ++c)
        async_b128(ldsB0 + c * 16, (const char*)gB + c * 16);

    // ---- stage 64x512 fp32 values tile -> f16 LDS tile ----
    {
        const int c4 = tid & 127;        // float4 column
        const int r0 = tid >> 7;         // 0..3
        #pragma unroll
        for (int rr = 0; rr < 16; ++rr) {
            const int row = r0 + rr * 4;
            const float4 v = *(const float4*)(values +
                ((size_t)(b * TT + t0 + row) * HH) + c4 * 4);
            v4h h = { (_Float16)v.x, (_Float16)v.y, (_Float16)v.z, (_Float16)v.w };
            *(lv4h*)(sV + row * PITCH + c4 * 4) = h;    // ds_store_b64
        }
    }

    const int wave   = tid >> 5;
    const int lane   = tid & 31;
    const int ln     = lane & 15;
    const int laneHi = lane >> 4;
    const int mt     = wave >> 2;        // M-tile (16 rows)
    const int wn     = wave & 3;         // N chunk of 128 cols

    v8f acc[8];
    #pragma unroll
    for (int i = 0; i < 8; ++i) acc[i] = (v8f)(0.0f);

    const lh* aBase = sV + (mt * 16 + ln) * PITCH;

    #pragma unroll 1
    for (int j = 0; j < 8; ++j) {
        PROC_HALF(2 * j,     sB0, 1,       ldsB1);   // even slab from buf0
        PROC_HALF(2 * j + 1, sB1, (j < 7), ldsB0);   // odd slab from buf1
    }

    // ---- epilogue: tanh(acc + pqb[n]) * V[n], reduce over n ----
    float part[8];
    #pragma unroll
    for (int r = 0; r < 8; ++r) part[r] = 0.0f;

    #pragma unroll
    for (int nt = 0; nt < 8; ++nt) {
        const int n = wn * 128 + nt * 16 + ln;
        const float pq = pqb[b * UU + n];
        const float vv = Vv[n];
        #pragma unroll
        for (int r = 0; r < 8; ++r)
            part[r] += tanh_fast(acc[nt][r] + pq) * vv;
    }

    #pragma unroll
    for (int r = 0; r < 8; ++r) {
        float s = part[r];
        s += __shfl_xor(s, 1, 32);
        s += __shfl_xor(s, 2, 32);
        s += __shfl_xor(s, 4, 32);
        s += __shfl_xor(s, 8, 32);          // sum within 16-lane half
        if (ln == 0)
            atomicAdd(&sScorG[mt * 16 + laneHi * 8 + r], s);  // ds_add_f32
    }
    __syncthreads();

    if (tid < 64)
        scoreWS[(size_t)b * TT + t0 + tid] = sScorG[tid];
}

// ---------------------------------------------------------------------------
// Kernel 4: softmax over T per batch; write weights; zero context slice
// ---------------------------------------------------------------------------
__global__ void k_softmax(const float* __restrict__ scoreWS, float* __restrict__ out) {
    __shared__ float red[256];
    const int b = blockIdx.x;
    const int tid = threadIdx.x;
    const int WOFF = BB * HH;

    float sc[8];
    float m = -1e30f;
    #pragma unroll
    for (int j = 0; j < 8; ++j) {
        sc[j] = scoreWS[(size_t)b * TT + tid + j * 256];
        m = fmaxf(m, sc[j]);
    }
    red[tid] = m; __syncthreads();
    for (int s = 128; s > 0; s >>= 1) {
        if (tid < s) red[tid] = fmaxf(red[tid], red[tid + s]);
        __syncthreads();
    }
    m = red[0]; __syncthreads();

    float e = 0.0f;
    #pragma unroll
    for (int j = 0; j < 8; ++j) e += __expf(sc[j] - m);
    red[tid] = e; __syncthreads();
    for (int s = 128; s > 0; s >>= 1) {
        if (tid < s) red[tid] += red[tid + s];
        __syncthreads();
    }
    const float inv = 1.0f / red[0];

    #pragma unroll
    for (int j = 0; j < 8; ++j)
        out[WOFF + (size_t)b * TT + tid + j * 256] = __expf(sc[j] - m) * inv;

    // zero the context slice this batch owns (before k_context atomics)
    for (int h = tid; h < HH; h += 256) out[b * HH + h] = 0.0f;
}

// ---------------------------------------------------------------------------
// Kernel 5: context[b,h] += sum_t w[b,t]*values[b,t,h]  (split over t-chunks)
// ---------------------------------------------------------------------------
__global__ void k_context(const float* __restrict__ values, float* __restrict__ out) {
    __shared__ float sw[256];
    const int tc = blockIdx.x;
    const int b  = blockIdx.y;
    const int tid = threadIdx.x;
    const int WOFF = BB * HH;

    sw[tid] = out[WOFF + (size_t)b * TT + tc * 256 + tid];
    __syncthreads();

    float a0 = 0.0f, a1 = 0.0f;
    for (int t2 = 0; t2 < 256; ++t2) {
        const float w = sw[t2];
        const float* vr = values + (size_t)(b * TT + tc * 256 + t2) * HH;
        a0 += w * vr[tid];
        a1 += w * vr[tid + 256];
    }
    atomicAdd(&out[b * HH + tid],        a0);
    atomicAdd(&out[b * HH + tid + 256],  a1);
}

// ---------------------------------------------------------------------------
extern "C" void kernel_launch(void* const* d_in, const int* in_sizes, int n_in,
                              void* d_out, int out_size, void* d_ws, size_t ws_size,
                              hipStream_t stream) {
    const float* query     = (const float*)d_in[0];
    const float* cellstate = (const float*)d_in[1];
    const float* values    = (const float*)d_in[2];
    const float* W1        = (const float*)d_in[3];
    const float* b1        = (const float*)d_in[4];
    const float* W2        = (const float*)d_in[5];
    const float* b2        = (const float*)d_in[6];
    const float* Vv        = (const float*)d_in[7];
    float* out = (float*)d_out;

    // workspace layout
    char* ws = (char*)d_ws;
    float*    pqb     = (float*)ws;                              // 64*512 f32
    _Float16* W2t     = (_Float16*)(ws + BB * UU * 4);           // 512*512 f16
    float*    scoreWS = (float*)(ws + BB * UU * 4 + HH * UU * 2);// 64*2048 f32

    k_projq<<<dim3(BB), dim3(512), 0, stream>>>(query, cellstate, W1, b1, b2, pqb);
    k_packW2<<<dim3((HH * UU) / 256), dim3(256), 0, stream>>>(W2, W2t);
    k_score<<<dim3(TT / 64, BB), dim3(512), SMEM_K3, stream>>>(values, W2t, pqb, Vv, scoreWS);
    k_softmax<<<dim3(BB), dim3(256), 0, stream>>>(scoreWS, out);
    k_context<<<dim3(8, BB), dim3(256), 0, stream>>>(values, out);
}
